// RQVAE_8400956031136
// MI455X (gfx1250) — matrix operand order repaired and checked by reading, not compiled, not run on hardware
//
#include <hip/hip_runtime.h>

typedef __attribute__((ext_vector_type(16))) __bf16 v16bf;
typedef __attribute__((ext_vector_type(8)))  float  v8f;

#define NB 65536L   // batch
#define DIN 768
#define DLAT 32
#define NCB 3
#define CBK 256

// ---------- helpers ----------
__device__ __forceinline__ unsigned short f2bf(float f) {
    unsigned u = __builtin_bit_cast(unsigned, f);
    u += 0x7FFFu + ((u >> 16) & 1u);          // round-to-nearest-even
    return (unsigned short)(u >> 16);
}
__device__ __forceinline__ float bf2f(unsigned short s) {
    unsigned u = ((unsigned)s) << 16;
    return __builtin_bit_cast(float, u);
}
// A-fragment dim map for 16-bit 16x32 A tile (ISA 7.12.2):
// lane half h, vector element j -> K index
__device__ __forceinline__ int dmap(int j, int h) {
    int i = j >> 1, s = j & 1;
    return (i < 4) ? (8 * h + 2 * i + s) : (16 + 8 * h + 2 * (i - 4) + s);
}

// ---------- conversion kernels (weights / codebooks only; x is fused) ----------
__global__ void k_f32_to_bf16(const float* __restrict__ in,
                              unsigned short* __restrict__ out, long n) {
    long i = (long)blockIdx.x * blockDim.x + threadIdx.x;
    long st = (long)gridDim.x * blockDim.x;
    for (; i < n; i += st) out[i] = f2bf(in[i]);
}

__global__ void k_cbnorm(const float* __restrict__ cb, float* __restrict__ cbn) {
    int i = blockIdx.x * blockDim.x + threadIdx.x;  // 768 codewords total
    if (i < NCB * CBK) {
        float s = 0.f;
        #pragma unroll
        for (int d = 0; d < DLAT; d++) { float v = cb[i * DLAT + d]; s += v * v; }
        cbn[i] = s;
    }
}

// ---------- tiled WMMA GEMM: C[M x N] = act(A @ W_bf16 + bias) ----------
// block: 256 thr = 8 waves; block tile 128x32; each wave owns two 16x16 accs
template <bool AF32, bool RELU, bool SBF, bool SF32>
__global__ __launch_bounds__(256) void k_gemm_bf16(
    const unsigned short* __restrict__ A16, const float* __restrict__ A32,
    const unsigned short* __restrict__ W, const float* __restrict__ bias,
    unsigned short* __restrict__ Cb, float* __restrict__ Cf, int N, int K)
{
    __shared__ unsigned lAu[128 * 32 / 2];                // A tile 128x32 bf16
    __shared__ __align__(8) unsigned short lW[32 * 32];   // W tile transposed [n][k]

    int tid = threadIdx.x;
    int wave = tid >> 5, lane = tid & 31;
    int wr = wave & 3, wc = wave >> 2;
    int h = lane >> 4, l16 = lane & 15;
    long rowBase = (long)blockIdx.x * 128;
    int  colBase = blockIdx.y * 32;

    v8f acc0 = {}, acc1 = {};
    for (int kt = 0; kt < K; kt += 32) {
        // stage A: 2048 u32 (128 rows x 16 u32), coalesced; fused f32->bf16 if AF32
        #pragma unroll
        for (int q = 0; q < 8; q++) {
            int uu = tid + q * 256;               // u32 index in tile
            int r = uu >> 4, c2 = (uu & 15) * 2;  // row, even col
            long off = (rowBase + r) * (long)K + kt + c2;
            if (AF32) {
                float f0 = A32[off], f1 = A32[off + 1];
                lAu[uu] = (unsigned)f2bf(f0) | ((unsigned)f2bf(f1) << 16);
                if (kt + 32 < K) __builtin_prefetch(A32 + off + 32, 0, 3);
            } else {
                lAu[uu] = *(const unsigned*)(A16 + off);
                if (kt + 32 < K) __builtin_prefetch(A16 + off + 32, 0, 3);
            }
        }
        // stage W transposed: lW[n][k] = W[kt+k][colBase+n]
        {
            int n = tid & 31, k0 = (tid >> 5) * 4;
            #pragma unroll
            for (int q = 0; q < 4; q++)
                lW[n * 32 + k0 + q] = W[(long)(kt + k0 + q) * N + colBase + n];
        }
        __syncthreads();

        union { v16bf v; unsigned u[8]; } a0, a1, b;
        int arow0 = wr * 16 + l16;
        #pragma unroll
        for (int i = 0; i < 8; i++) {
            int du2 = (i < 4) ? (4 * h + i) : (4 + 4 * h + i); // dmap/2
            a0.u[i] = lAu[arow0 * 16 + du2];
            a1.u[i] = lAu[(64 + arow0) * 16 + du2];
        }
        int bn = wc * 16 + l16;
        const unsigned* lWu = (const unsigned*)lW;
        #pragma unroll
        for (int i = 0; i < 8; i++) b.u[i] = lWu[bn * 16 + ((i < 4) ? (4 * h + i) : (4 + 4 * h + i))];

        acc0 = __builtin_amdgcn_wmma_f32_16x16x32_bf16(
            false, a0.v, false, b.v, (short)0, acc0, false, false);
        acc1 = __builtin_amdgcn_wmma_f32_16x16x32_bf16(
            false, a1.v, false, b.v, (short)0, acc1, false, false);
        __syncthreads();
    }

    int col = colBase + wc * 16 + l16;
    float bc = bias[col];
    #pragma unroll
    for (int v = 0; v < 8; v++) {
        long row0 = rowBase + wr * 16 + v + 8 * h;
        long row1 = row0 + 64;
        float x0 = acc0[v] + bc;
        float x1 = acc1[v] + bc;
        if (RELU) { x0 = fmaxf(x0, 0.f); x1 = fmaxf(x1, 0.f); }
        if (SBF) {
            Cb[row0 * N + col] = f2bf(x0);
            Cb[row1 * N + col] = f2bf(x1);
        }
        if (SF32) {
            Cf[row0 * N + col] = x0;
            Cf[row1 * N + col] = x1;
        }
    }
}

// ---------- residual VQ: 3 levels, 256 codewords, WMMA distance search ----------
// block: 256 thr = 8 waves, wave handles 16 samples
__global__ __launch_bounds__(256) void k_rq_quant(
    const float* __restrict__ z_enc, const unsigned short* __restrict__ cb_bf,
    const float* __restrict__ cbn, unsigned short* __restrict__ zhat_bf,
    float* __restrict__ commit)
{
    __shared__ unsigned cbl[NCB * CBK * DLAT / 2];  // 48 KB bf16 codebooks
    __shared__ float    cbnl[NCB * CBK];            // 3 KB norms
    __shared__ unsigned idxs[8][16];
    __shared__ float    snrm[8][16];

    int tid = threadIdx.x;
    for (int i = tid; i < NCB * CBK * DLAT / 2; i += 256)
        cbl[i] = ((const unsigned*)cb_bf)[i];
    for (int i = tid; i < NCB * CBK; i += 256) cbnl[i] = cbn[i];

    int wave = tid >> 5, lane = tid & 31, h = lane >> 4, l16 = lane & 15;
    if (lane == 0)  { for (int v = 0; v < 8; v++) snrm[wave][v] = 0.f; }
    if (lane == 16) { for (int v = 0; v < 8; v++) snrm[wave][v + 8] = 0.f; }
    __syncthreads();

    long base = ((long)blockIdx.x * 8 + wave) * 16;
    long row = base + l16;
    float r[16], z0[16];
    #pragma unroll
    for (int j = 0; j < 16; j++) { z0[j] = r[j] = z_enc[row * DLAT + dmap(j, h)]; }

    const unsigned short* cbs = (const unsigned short*)cbl;

    for (int lev = 0; lev < NCB; lev++) {
        union { v16bf v; unsigned u[8]; } a;
        #pragma unroll
        for (int i = 0; i < 8; i++)
            a.u[i] = (unsigned)f2bf(r[2 * i]) | ((unsigned)f2bf(r[2 * i + 1]) << 16);

        float best[8]; int bidx[8];
        #pragma unroll
        for (int v = 0; v < 8; v++) { best[v] = 3.4e38f; bidx[v] = 0; }

        const unsigned* cbase = cbl + lev * CBK * DLAT / 2;
        const float* nlev = cbnl + lev * CBK;
        for (int nb = 0; nb < 16; nb++) {
            union { v16bf v; unsigned u[8]; } b;
            int n = nb * 16 + l16;
            #pragma unroll
            for (int i = 0; i < 8; i++) b.u[i] = cbase[n * 16 + ((i < 4) ? (4 * h + i) : (4 + 4 * h + i))];
            v8f d = {};
            d = __builtin_amdgcn_wmma_f32_16x16x32_bf16(
                false, a.v, false, b.v, (short)0, d, false, false);
            #pragma unroll
            for (int v = 0; v < 8; v++) {
                float sc = nlev[n] - 2.f * d[v];   // argmin of ||c||^2 - 2 r.c
                bool take = sc < best[v];
                best[v] = take ? sc : best[v];
                bidx[v] = take ? n : bidx[v];
            }
        }
        // branchless reduce over the 16 lanes of each half
        // (rows 0-7 live in half0 lanes, rows 8-15 in half1 lanes)
        #pragma unroll
        for (int off = 8; off >= 1; off >>= 1) {
            #pragma unroll
            for (int v = 0; v < 8; v++) {
                float os = __shfl_xor(best[v], off, 32);
                int   oi = __shfl_xor(bidx[v], off, 32);
                // non-short-circuit logic -> pure v_cmp/v_cndmask, no exec branches
                int take = (int)(os < best[v]) | ((int)(os == best[v]) & (int)(oi < bidx[v]));
                best[v] = take ? os : best[v];
                bidx[v] = take ? oi : bidx[v];
            }
        }
        if (lane == 0)
            for (int v = 0; v < 8; v++) { idxs[wave][v] = bidx[v];     snrm[wave][v]     += cbnl[lev * CBK + bidx[v]]; }
        if (lane == 16)
            for (int v = 0; v < 8; v++) { idxs[wave][v + 8] = bidx[v]; snrm[wave][v + 8] += cbnl[lev * CBK + bidx[v]]; }
        __syncthreads();
        unsigned my = idxs[wave][l16];
        #pragma unroll
        for (int j = 0; j < 16; j++)
            r[j] -= bf2f(cbs[lev * CBK * DLAT + my * DLAT + dmap(j, h)]);
        __syncthreads();
    }

    float pd = 0.f, pr = 0.f;
    #pragma unroll
    for (int j = 0; j < 16; j++) {
        float zh = z0[j] - r[j];
        pd += zh * r[j];
        pr += r[j] * r[j];
        zhat_bf[row * DLAT + dmap(j, h)] = f2bf(zh);
    }
    pd += __shfl_xor(pd, 16, 32);
    pr += __shfl_xor(pr, 16, 32);
    // writer lanes: 0..7 (rows base+0..7), 24..31 (rows base+8..15)
    if (lane < 8 || lane >= 24) {
        float sn = snrm[wave][l16];
        commit[base + l16] = 1.25f * (sn + 3.f * pr - 2.f * pd);
    }
}

// ---------- per-sample recon + commit ----------
__global__ __launch_bounds__(256) void k_recon(
    const float* __restrict__ xhat, const float* __restrict__ x,
    const float* __restrict__ commit, float* __restrict__ persample)
{
    __shared__ float red[256];
    long rowb = (long)blockIdx.x * DIN;
    float s = 0.f;
    for (int c = threadIdx.x; c < DIN; c += 256) {
        float d = xhat[rowb + c] - x[rowb + c];
        s += d * d;
    }
    red[threadIdx.x] = s; __syncthreads();
    for (int off = 128; off; off >>= 1) {
        if (threadIdx.x < off) red[threadIdx.x] += red[threadIdx.x + off];
        __syncthreads();
    }
    if (!threadIdx.x) persample[blockIdx.x] = red[0] + commit[blockIdx.x];
}

__global__ __launch_bounds__(256) void k_reduce1(const float* __restrict__ in,
                                                 float* __restrict__ out) {
    __shared__ float red[256];
    red[threadIdx.x] = in[(long)blockIdx.x * 256 + threadIdx.x];
    __syncthreads();
    for (int off = 128; off; off >>= 1) {
        if (threadIdx.x < off) red[threadIdx.x] += red[threadIdx.x + off];
        __syncthreads();
    }
    if (!threadIdx.x) out[blockIdx.x] = red[0];
}
__global__ __launch_bounds__(256) void k_reduce2(const float* __restrict__ in,
                                                 float* __restrict__ loss) {
    __shared__ float red[256];
    red[threadIdx.x] = in[threadIdx.x];
    __syncthreads();
    for (int off = 128; off; off >>= 1) {
        if (threadIdx.x < off) red[threadIdx.x] += red[threadIdx.x + off];
        __syncthreads();
    }
    if (!threadIdx.x) *loss = red[0] * (1.0f / 65536.0f);
}

// ---------- launch ----------
extern "C" void kernel_launch(void* const* d_in, const int* in_sizes, int n_in,
                              void* d_out, int out_size, void* d_ws, size_t ws_size,
                              hipStream_t stream) {
    const float* x  = (const float*)d_in[0];
    const float* ew[4] = {(const float*)d_in[1], (const float*)d_in[2],
                          (const float*)d_in[3], (const float*)d_in[4]};
    const float* eb[4] = {(const float*)d_in[5], (const float*)d_in[6],
                          (const float*)d_in[7], (const float*)d_in[8]};
    const float* dw[4] = {(const float*)d_in[9],  (const float*)d_in[10],
                          (const float*)d_in[11], (const float*)d_in[12]};
    const float* db[4] = {(const float*)d_in[13], (const float*)d_in[14],
                          (const float*)d_in[15], (const float*)d_in[16]};
    const float* cb = (const float*)d_in[17];

    float* xhat = (float*)d_out;
    float* loss = xhat + NB * DIN;

    const int ENC[5] = {768, 512, 256, 128, 32};
    const int DEC[5] = {32, 128, 256, 512, 768};

    char* w = (char*)d_ws;
    size_t o = 0;
    auto alloc = [&](size_t bytes) -> char* {
        size_t r = (o + 255) & ~(size_t)255;
        o = r + bytes;
        return w + r;
    };
    unsigned short* h1   = (unsigned short*)alloc(NB * 512 * 2);
    unsigned short* h2   = (unsigned short*)alloc(NB * 256 * 2);
    unsigned short* h3   = (unsigned short*)alloc(NB * 128 * 2);
    float*          zenc = (float*)alloc(NB * DLAT * 4);
    unsigned short* zbf  = (unsigned short*)alloc(NB * DLAT * 2);
    float*          comm = (float*)alloc(NB * 4);
    float*          pers = (float*)alloc(NB * 4);
    float*          part = (float*)alloc(256 * 4);
    unsigned short* ewb[4]; unsigned short* dwb[4];
    for (int i = 0; i < 4; i++) ewb[i] = (unsigned short*)alloc((size_t)ENC[i] * ENC[i + 1] * 2);
    for (int i = 0; i < 4; i++) dwb[i] = (unsigned short*)alloc((size_t)DEC[i] * DEC[i + 1] * 2);
    unsigned short* cbbf = (unsigned short*)alloc((size_t)NCB * CBK * DLAT * 2);
    float*          cbn  = (float*)alloc((size_t)NCB * CBK * 4);
    (void)ws_size; (void)in_sizes; (void)n_in; (void)out_size;

    // weight / codebook conversions
    for (int i = 0; i < 4; i++) {
        long n = (long)ENC[i] * ENC[i + 1];
        k_f32_to_bf16<<<(unsigned)((n + 255) / 256), 256, 0, stream>>>(ew[i], ewb[i], n);
        n = (long)DEC[i] * DEC[i + 1];
        k_f32_to_bf16<<<(unsigned)((n + 255) / 256), 256, 0, stream>>>(dw[i], dwb[i], n);
    }
    k_f32_to_bf16<<<96, 256, 0, stream>>>(cb, cbbf, (long)NCB * CBK * DLAT);
    k_cbnorm<<<3, 256, 0, stream>>>(cb, cbn);

    // encoder (layer 1 fuses x f32->bf16 in the A staging path)
    {
        dim3 g((unsigned)(NB / 128), (unsigned)(ENC[1] / 32));
        k_gemm_bf16<true, true, true, false><<<g, 256, 0, stream>>>(
            nullptr, x, ewb[0], eb[0], h1, nullptr, ENC[1], ENC[0]);
    }
    {
        dim3 g((unsigned)(NB / 128), (unsigned)(ENC[2] / 32));
        k_gemm_bf16<false, true, true, false><<<g, 256, 0, stream>>>(
            h1, nullptr, ewb[1], eb[1], h2, nullptr, ENC[2], ENC[1]);
    }
    {
        dim3 g((unsigned)(NB / 128), (unsigned)(ENC[3] / 32));
        k_gemm_bf16<false, true, true, false><<<g, 256, 0, stream>>>(
            h2, nullptr, ewb[2], eb[2], h3, nullptr, ENC[3], ENC[2]);
    }
    {
        dim3 g((unsigned)(NB / 128), (unsigned)(ENC[4] / 32));
        k_gemm_bf16<false, false, false, true><<<g, 256, 0, stream>>>(
            h3, nullptr, ewb[3], eb[3], nullptr, zenc, ENC[4], ENC[3]);
    }

    // residual VQ
    k_rq_quant<<<(unsigned)(NB / 128), 256, 0, stream>>>(zenc, cbbf, cbn, zbf, comm);

    // decoder
    {
        dim3 g((unsigned)(NB / 128), (unsigned)(DEC[1] / 32));
        k_gemm_bf16<false, true, true, false><<<g, 256, 0, stream>>>(
            zbf, nullptr, dwb[0], db[0], h3, nullptr, DEC[1], DEC[0]);
    }
    {
        dim3 g((unsigned)(NB / 128), (unsigned)(DEC[2] / 32));
        k_gemm_bf16<false, true, true, false><<<g, 256, 0, stream>>>(
            h3, nullptr, dwb[1], db[1], h2, nullptr, DEC[2], DEC[1]);
    }
    {
        dim3 g((unsigned)(NB / 128), (unsigned)(DEC[3] / 32));
        k_gemm_bf16<false, true, true, false><<<g, 256, 0, stream>>>(
            h2, nullptr, dwb[2], db[2], h1, nullptr, DEC[3], DEC[2]);
    }
    {
        dim3 g((unsigned)(NB / 128), (unsigned)(DEC[4] / 32));
        k_gemm_bf16<false, false, false, true><<<g, 256, 0, stream>>>(
            h1, nullptr, dwb[3], db[3], nullptr, xhat, DEC[4], DEC[3]);
    }

    // loss
    k_recon<<<(unsigned)NB, 256, 0, stream>>>(xhat, x, comm, pers);
    k_reduce1<<<256, 256, 0, stream>>>(pers, part);
    k_reduce2<<<1, 256, 0, stream>>>(part, loss);
}
